// RNNDecoder_multi_88493506167466
// MI455X (gfx1250) — matrix operand order repaired and checked by reading, not compiled, not run on hardware
//
#include <hip/hip_runtime.h>
#include <math.h>

// ---------------------------------------------------------------------------
// RNN relational decoder for MI455X (gfx1250), wave32 + WMMA f16->f32.
//
// Shapes: B=16, A=32, T=12, F=4, H=256, K=4, C=16, E=992.
// Edge GEMMs (M=15872) dominate (~25 GFLOP/step, 11 sequential steps).
// All weights/activations are L2-resident (192MB), so the bottleneck is
// matrix-pipe feed rate: WMMA tiles are staged through LDS so each
// 64x32-block tile is fetched from L2 once and reused by all 8 waves
// (A tile shared by 2 N-waves, B tile by 4 M-waves: ~2.7x less L2 traffic).
// Global->LDS staging uses gfx1250 async-LDS copies (ASYNCcnt +
// s_wait_asynccnt) when the builtins are available, else a VGPR bounce.
// ---------------------------------------------------------------------------

#define B_ 16
#define A_ 32
#define T_ 12
#define F_ 4
#define H_ 256
#define K_ 4
#define C_ 16
#define E_ (A_ * (A_ - 1))   // 992 edges
#define ME (B_ * E_)         // 15872 edge rows (multiple of 64)
#define MN (B_ * A_)         // 512 node rows  (multiple of 64)

// Padded LDS row stride: 40 halves = 80B = 20 banks -> conflict-free b128
// fragment reads across the 16 row-lanes, and keeps 16B chunk alignment.
#define LDS_STRIDE 40

typedef __attribute__((ext_vector_type(16))) _Float16 v16h;
typedef __attribute__((ext_vector_type(8)))  float    v8f;
typedef __attribute__((ext_vector_type(4)))  unsigned int u32x4;
typedef __attribute__((ext_vector_type(4)))  int      i32x4;

#if defined(__gfx1250__) && __has_builtin(__builtin_amdgcn_global_load_async_to_lds_b128) && \
    __has_builtin(__builtin_amdgcn_s_wait_asynccnt)
#define USE_ASYNC_LDS 1
#else
#define USE_ASYNC_LDS 0
#endif

// 16-byte global -> LDS copy (async path emits global_load_async_to_lds_b128).
// Builtin signature (from probe diagnostic): arg0 is AS1 int4*, arg1 AS3 int4*.
__device__ __forceinline__ void g2lds16(const _Float16* __restrict__ g,
                                        _Float16* __restrict__ l) {
#if USE_ASYNC_LDS
  void* gv = (void*)g;  // drop const, then addrspace-qualify
  void* lv = (void*)l;
  __builtin_amdgcn_global_load_async_to_lds_b128(
      (__attribute__((address_space(1))) i32x4*)gv,
      (__attribute__((address_space(3))) i32x4*)lv, 0, 0);
#else
  *(u32x4*)l = *(const u32x4*)g;
#endif
}

__device__ __forceinline__ void g2lds_join() {
#if USE_ASYNC_LDS
  __builtin_amdgcn_s_wait_asynccnt(0);
#endif
}

// ---------------------------------------------------------------------------
// WMMA fragment loaders from staged LDS tiles (ISA 7.12.2 layouts, wave32).
// A (16x32 f16): lane l holds row m = l&15; K-halves by l>>4:
//   halves 0..7 -> K = 8*(l>>4)+0..7 ; halves 8..15 -> K = 16+8*(l>>4)+0..7
// B (32x16 f16 = W^T from row-major W): lane l holds col n = l&15,
//   same K order as A: halves 0..7 -> K = 8*(l>>4)+0..7 ; 8..15 -> K = 16+8*(l>>4)+0..7
// C/D (16x16 f32): VGPR r -> row = r + 8*(l>>4), col = l&15.
// ---------------------------------------------------------------------------
__device__ __forceinline__ v16h lds_a_frag(const _Float16* sA, int mLoc, int lane) {
  const int m  = mLoc + (lane & 15);
  const int kh = lane >> 4;
  const _Float16* p = sA + m * LDS_STRIDE + kh * 8;
  v16h a;
#pragma unroll
  for (int i = 0; i < 8; ++i) a[i] = p[i];
#pragma unroll
  for (int i = 0; i < 8; ++i) a[8 + i] = p[16 + i];
  return a;
}

__device__ __forceinline__ v16h lds_b_frag(const _Float16* sB, int nLoc, int lane) {
  const int n  = nLoc + (lane & 15);
  const int k0 = (lane >> 4) << 3;
  const _Float16* p = sB + n * LDS_STRIDE + k0;
  v16h b;
#pragma unroll
  for (int i = 0; i < 8; ++i) b[i] = p[i];
#pragma unroll
  for (int i = 0; i < 8; ++i) b[8 + i] = p[16 + i];
  return b;
}

// ---------------------------------------------------------------------------
// Generic LDS-staged WMMA GEMM: Y = act(X @ W^T + bias)
//   X: [M x KD] f16 row-major, W: [N x KD] f16 row-major (used as B = W^T)
//   blockDim = 256 (8 waves, 4Mx2N), block tile 64x32, wave tile 16x16.
//   blockIdx.z batches over independent problems via strides (K=4 edge types).
// ---------------------------------------------------------------------------
template <int KD>
__global__ void wmma_gemm_kernel(const _Float16* __restrict__ X, size_t sX,
                                 const _Float16* __restrict__ W, size_t sW,
                                 const float* __restrict__ bias, size_t sB,
                                 _Float16* __restrict__ Yh, float* __restrict__ Yf,
                                 size_t sY, int N, int doRelu) {
  __shared__ __align__(16) _Float16 sAt[64 * LDS_STRIDE];
  __shared__ __align__(16) _Float16 sBt[32 * LDS_STRIDE];

  const int tid  = threadIdx.x;
  const int lane = tid & 31;
  const int wave = tid >> 5;
  const int z    = blockIdx.z;
  X += (size_t)z * sX;
  W += (size_t)z * sW;
  const float* bz = bias ? bias + (size_t)z * sB : nullptr;

  const int mBlk = blockIdx.x * 64;
  const int nBlk = blockIdx.y * 32;
  const int mLoc = (wave & 3) * 16;
  const int nLoc = (wave >> 2) * 16;

  // cooperative stage mapping: 16B chunks
  const int rowA = tid >> 2, chA = tid & 3;           // 64 rows x 4 chunks
  const int rowB = (tid & 127) >> 2, chB = tid & 3;   // 32 rows x 4 chunks

  v8f c = {};
#pragma unroll
  for (int kk = 0; kk < KD; kk += 32) {
    __syncthreads();  // previous-iteration readers done
    g2lds16(&X[(size_t)(mBlk + rowA) * KD + kk + chA * 8],
            &sAt[rowA * LDS_STRIDE + chA * 8]);
    if (tid < 128)
      g2lds16(&W[(size_t)(nBlk + rowB) * KD + kk + chB * 8],
              &sBt[rowB * LDS_STRIDE + chB * 8]);
    if (kk + 32 < KD)  // stream next tile toward L2 -> global_prefetch_b8
      __builtin_prefetch(&W[(size_t)(nBlk + rowB) * KD + kk + 32], 0, 1);
    g2lds_join();
    __syncthreads();  // staged tile visible to all waves
    v16h a = lds_a_frag(sAt, mLoc, lane);
    v16h b = lds_b_frag(sBt, nLoc, lane);
    c = __builtin_amdgcn_wmma_f32_16x16x32_f16(false, a, false, b, (short)0, c,
                                               false, false);
  }

  const int   col    = nBlk + nLoc + (lane & 15);
  const int   rowOff = (lane >> 4) << 3;
  const float bval   = bz ? bz[col] : 0.0f;
#pragma unroll
  for (int r = 0; r < 8; ++r) {
    float v = c[r] + bval;
    if (doRelu) v = v > 0.0f ? v : 0.0f;
    const size_t row = (size_t)(mBlk + mLoc + r + rowOff);
    if (Yh) Yh[(size_t)z * sY + row * N + col] = (_Float16)v;
    if (Yf) Yf[(size_t)z * sY + row * N + col] = v;
  }
}

// ---------------------------------------------------------------------------
// Specialized second edge GEMM, fused with bias+ReLU+rel_type weighting:
//   all_msgs[row,n] = sum_k rel_type[row,k] * relu(h1[k] @ W2[k]^T + b2[k])[row,n]
// h2 never hits memory: the 4 per-k accumulators are combined in VGPRs.
// ---------------------------------------------------------------------------
__global__ void edge_msg2_kernel(const _Float16* __restrict__ h1,   // [K][ME][H]
                                 const _Float16* __restrict__ W2h,  // [K][H][H]
                                 const float* __restrict__ b2,      // [K][H]
                                 const float* __restrict__ relt,    // [ME][K]
                                 float* __restrict__ msgs) {        // [ME][H]
  __shared__ __align__(16) _Float16 sAt[64 * LDS_STRIDE];
  __shared__ __align__(16) _Float16 sBt[32 * LDS_STRIDE];

  const int tid  = threadIdx.x;
  const int lane = tid & 31;
  const int wave = tid >> 5;
  const int mBlk = blockIdx.x * 64;
  const int nBlk = blockIdx.y * 32;
  const int mLoc = (wave & 3) * 16;
  const int nLoc = (wave >> 2) * 16;
  const int col    = nBlk + nLoc + (lane & 15);
  const int rowOff = (lane >> 4) << 3;

  const int rowA = tid >> 2, chA = tid & 3;
  const int rowB = (tid & 127) >> 2, chB = tid & 3;

  float acc[8];
#pragma unroll
  for (int r = 0; r < 8; ++r) acc[r] = 0.0f;

#pragma unroll
  for (int k = 0; k < K_; ++k) {
    const _Float16* Xk = h1 + (size_t)k * ME * H_;
    const _Float16* Wk = W2h + (size_t)k * H_ * H_;
    v8f c = {};
#pragma unroll
    for (int kk = 0; kk < H_; kk += 32) {
      __syncthreads();
      g2lds16(&Xk[(size_t)(mBlk + rowA) * H_ + kk + chA * 8],
              &sAt[rowA * LDS_STRIDE + chA * 8]);
      if (tid < 128)
        g2lds16(&Wk[(size_t)(nBlk + rowB) * H_ + kk + chB * 8],
                &sBt[rowB * LDS_STRIDE + chB * 8]);
      g2lds_join();
      __syncthreads();
      v16h a = lds_a_frag(sAt, mLoc, lane);
      v16h b = lds_b_frag(sBt, nLoc, lane);
      c = __builtin_amdgcn_wmma_f32_16x16x32_f16(false, a, false, b, (short)0, c,
                                                 false, false);
    }
    const float bval = b2[k * H_ + col];
#pragma unroll
    for (int r = 0; r < 8; ++r) {
      float v = c[r] + bval;
      v = v > 0.0f ? v : 0.0f;  // relu(h2)
      const int row = mBlk + mLoc + r + rowOff;
      acc[r] += relt[(size_t)row * K_ + k] * v;
    }
  }
#pragma unroll
  for (int r = 0; r < 8; ++r) {
    const int row = mBlk + mLoc + r + rowOff;
    msgs[(size_t)row * H_ + col] = acc[r];
  }
}

// ---------------------------------------------------------------------------
// Elementwise / small kernels
// ---------------------------------------------------------------------------

// pre_msg[b,e,:] = concat(hidden[b, recv(e)], hidden[b, send(e)]) in f16.
// recv(e)=e/31 ; send(e)=r+(r>=i) with r=e%31 (one-hot rel_rec/rel_send analytic).
__global__ void premsg_kernel(const _Float16* __restrict__ hid_h,
                              _Float16* __restrict__ pm) {
  const int idx = blockIdx.x * blockDim.x + threadIdx.x;  // ME * 2H
  if (idx >= ME * 2 * H_) return;
  const int h   = idx & (2 * H_ - 1);
  const int row = idx >> 9;  // /512
  const int b = row / E_, e = row % E_;
  const int i = e / (A_ - 1), rr = e % (A_ - 1);
  const int j = rr + (rr >= i ? 1 : 0);
  const int node = (h < H_) ? i : j;
  pm[idx] = hid_h[((size_t)(b * A_ + node)) * H_ + (h & (H_ - 1))];
}

// agg[b,a,:] = sum over the 31 contiguous edges with receiver a  + cln_msg, -> f16
__global__ void aggregate_kernel(const float* __restrict__ msgs,
                                 const float* __restrict__ cln,
                                 _Float16* __restrict__ aggh) {
  const int idx = blockIdx.x * blockDim.x + threadIdx.x;  // MN * H
  if (idx >= MN * H_) return;
  const int h  = idx & (H_ - 1);
  const int na = idx >> 8;
  const int b = na / A_, a = na % A_;
  const float* p = msgs + ((size_t)b * E_ + (size_t)a * (A_ - 1)) * H_ + h;
  float s = cln[idx];
  for (int r = 0; r < A_ - 1; ++r) s += p[(size_t)r * H_];
  aggh[idx] = (_Float16)s;
}

// Fused GRU cell update. The F=4 input matvecs are tiny -> per-thread.
__global__ void gru_kernel(const float* __restrict__ x,       // [MN][F]
                           const float* __restrict__ hr_lin,  // agg@W_hr^T+b_hr
                           const float* __restrict__ hn_lin,  // agg@W_hn^T+b_hn
                           const float* __restrict__ Wir, const float* __restrict__ bir,
                           const float* __restrict__ Wiz, const float* __restrict__ biz,
                           const float* __restrict__ Win, const float* __restrict__ bin_,
                           float* __restrict__ hid, _Float16* __restrict__ hid_h) {
  const int idx = blockIdx.x * blockDim.x + threadIdx.x;  // MN * H
  if (idx >= MN * H_) return;
  const int h  = idx & (H_ - 1);
  const int na = idx >> 8;
  const float* xv = x + (size_t)na * F_;
  float xr = bir[h], xz = biz[h], xn = bin_[h];
#pragma unroll
  for (int f = 0; f < F_; ++f) {
    xr += xv[f] * Wir[h * F_ + f];
    xz += xv[f] * Wiz[h * F_ + f];
    xn += xv[f] * Win[h * F_ + f];
  }
  const float hra = hr_lin[idx];
  const float r = 1.0f / (1.0f + expf(-(xr + hra)));
  const float z = 1.0f / (1.0f + expf(-(xz + hra)));
  const float n = tanhf(xn + r * hn_lin[idx]);
  const float hnew = (1.0f - z) * n + z * hid[idx];
  hid[idx]   = hnew;
  hid_h[idx] = (_Float16)hnew;
}

// pred = x + o2 @ W_out3^T + b_out3 ; write out[b,a,t,:] and advance x.
__global__ void pred_kernel(const _Float16* __restrict__ o2,
                            const float* __restrict__ W3, const float* __restrict__ b3,
                            float* __restrict__ x, float* __restrict__ out, int t) {
  const int idx = blockIdx.x * blockDim.x + threadIdx.x;  // MN * F
  if (idx >= MN * F_) return;
  const int f  = idx & 3;
  const int na = idx >> 2;
  const _Float16* ov = o2 + (size_t)na * H_;
  const float* w = W3 + (size_t)f * H_;
  float s = b3[f];
  for (int h = 0; h < H_; ++h) s += (float)ov[h] * w[h];
  const float p = x[idx] + s;
  x[idx] = p;
  out[((size_t)na * T_ + t) * F_ + f] = p;
}

// x0 = data[:, :, 0, :] ; also the t=0 slice of the output.
__global__ void init_x_kernel(const float* __restrict__ data,
                              float* __restrict__ x, float* __restrict__ out) {
  const int idx = blockIdx.x * blockDim.x + threadIdx.x;  // MN * F
  if (idx >= MN * F_) return;
  const int f  = idx & 3;
  const int na = idx >> 2;
  const float v = data[((size_t)na * T_) * F_ + f];
  x[idx] = v;
  out[((size_t)na * T_) * F_ + f] = v;
}

// Plain fp32 linear (used only for the tiny M=16 clinical MLPs).
__global__ void linear_f32_kernel(const float* __restrict__ X, const float* __restrict__ W,
                                  const float* __restrict__ bias, float* __restrict__ Y,
                                  int M, int N, int Kd, int doRelu) {
  const int idx = blockIdx.x * blockDim.x + threadIdx.x;
  if (idx >= M * N) return;
  const int n = idx % N, m = idx / N;
  const float* xr = X + (size_t)m * Kd;
  const float* wr = W + (size_t)n * Kd;
  float s = bias[n];
  for (int kd = 0; kd < Kd; ++kd) s += xr[kd] * wr[kd];
  if (doRelu && s < 0.0f) s = 0.0f;
  Y[idx] = s;
}

__global__ void cvt_f32_f16_kernel(const float* __restrict__ s,
                                   _Float16* __restrict__ d, int n) {
  const int idx = blockIdx.x * blockDim.x + threadIdx.x;
  if (idx < n) d[idx] = (_Float16)s[idx];
}

// ---------------------------------------------------------------------------
// Host launch sequence
// ---------------------------------------------------------------------------
extern "C" void kernel_launch(void* const* d_in, const int* in_sizes, int n_in,
                              void* d_out, int out_size, void* d_ws, size_t ws_size,
                              hipStream_t stream) {
  (void)in_sizes; (void)n_in; (void)out_size; (void)ws_size;

  const float* data     = (const float*)d_in[0];
  const float* rel_type = (const float*)d_in[1];
  // d_in[2]=rel_rec, d_in[3]=rel_send: one-hot eye rows, handled analytically.
  const float* clinical = (const float*)d_in[4];
  const float* W_msg1 = (const float*)d_in[5];
  const float* b_msg1 = (const float*)d_in[6];
  const float* W_msg2 = (const float*)d_in[7];
  const float* b_msg2 = (const float*)d_in[8];
  const float* W_in_r = (const float*)d_in[9];
  const float* b_in_r = (const float*)d_in[10];
  const float* W_in_z = (const float*)d_in[11];
  const float* b_in_z = (const float*)d_in[12];
  const float* W_in_n = (const float*)d_in[13];
  const float* b_in_n = (const float*)d_in[14];
  const float* W_hr   = (const float*)d_in[15];
  const float* b_hr   = (const float*)d_in[16];
  const float* W_hn   = (const float*)d_in[17];
  const float* b_hn   = (const float*)d_in[18];
  const float* W_out1 = (const float*)d_in[19];
  const float* b_out1 = (const float*)d_in[20];
  const float* W_out2 = (const float*)d_in[21];
  const float* b_out2 = (const float*)d_in[22];
  const float* W_out3 = (const float*)d_in[23];
  const float* b_out3 = (const float*)d_in[24];
  const float* Wc1 = (const float*)d_in[25];
  const float* bc1 = (const float*)d_in[26];
  const float* Wc2 = (const float*)d_in[27];
  const float* bc2 = (const float*)d_in[28];
  const float* Wc3 = (const float*)d_in[29];
  const float* bc3 = (const float*)d_in[30];
  const float* Wm1 = (const float*)d_in[31];
  const float* bm1 = (const float*)d_in[32];
  const float* Wm2 = (const float*)d_in[33];
  const float* bm2 = (const float*)d_in[34];
  const float* Wm3 = (const float*)d_in[35];
  const float* bm3 = (const float*)d_in[36];
  float* out = (float*)d_out;

  // --- workspace carve-out (256B aligned slabs) ---
  char* ws = (char*)d_ws;
  size_t off = 0;
  auto take = [&](size_t bytes) -> void* {
    void* p = ws + off;
    off = (off + bytes + 255) & ~(size_t)255;
    return p;
  };
  _Float16* W1h   = (_Float16*)take(sizeof(_Float16) * (size_t)K_ * H_ * 2 * H_);
  _Float16* W2h   = (_Float16*)take(sizeof(_Float16) * (size_t)K_ * H_ * H_);
  _Float16* Whr_h = (_Float16*)take(sizeof(_Float16) * (size_t)H_ * H_);
  _Float16* Whn_h = (_Float16*)take(sizeof(_Float16) * (size_t)H_ * H_);
  _Float16* Wo1_h = (_Float16*)take(sizeof(_Float16) * (size_t)H_ * H_);
  _Float16* Wo2_h = (_Float16*)take(sizeof(_Float16) * (size_t)H_ * H_);
  float*    hid   = (float*)take(sizeof(float) * (size_t)MN * H_);
  _Float16* hid_h = (_Float16*)take(sizeof(_Float16) * (size_t)MN * H_);
  float*    cln   = (float*)take(sizeof(float) * (size_t)MN * H_);
  float*    t1    = (float*)take(sizeof(float) * (size_t)B_ * H_);
  float*    t2    = (float*)take(sizeof(float) * (size_t)B_ * 2 * H_);
  float*    xcur  = (float*)take(sizeof(float) * (size_t)MN * F_);
  _Float16* pmsg  = (_Float16*)take(sizeof(_Float16) * (size_t)ME * 2 * H_);
  _Float16* h1    = (_Float16*)take(sizeof(_Float16) * (size_t)K_ * ME * H_);
  float*    msgs  = (float*)take(sizeof(float) * (size_t)ME * H_);
  _Float16* aggh  = (_Float16*)take(sizeof(_Float16) * (size_t)MN * H_);
  float*    hrl   = (float*)take(sizeof(float) * (size_t)MN * H_);
  float*    hnl   = (float*)take(sizeof(float) * (size_t)MN * H_);
  _Float16* o1    = (_Float16*)take(sizeof(_Float16) * (size_t)MN * H_);
  _Float16* o2    = (_Float16*)take(sizeof(_Float16) * (size_t)MN * H_);

  const int TB = 256;
  auto blocks = [](int n) { return (n + 255) / 256; };

  // --- one-time (per launch) f16 weight conversion; L2-resident thereafter ---
  cvt_f32_f16_kernel<<<blocks(K_ * H_ * 2 * H_), TB, 0, stream>>>(W_msg1, W1h, K_ * H_ * 2 * H_);
  cvt_f32_f16_kernel<<<blocks(K_ * H_ * H_), TB, 0, stream>>>(W_msg2, W2h, K_ * H_ * H_);
  cvt_f32_f16_kernel<<<blocks(H_ * H_), TB, 0, stream>>>(W_hr, Whr_h, H_ * H_);
  cvt_f32_f16_kernel<<<blocks(H_ * H_), TB, 0, stream>>>(W_hn, Whn_h, H_ * H_);
  cvt_f32_f16_kernel<<<blocks(H_ * H_), TB, 0, stream>>>(W_out1, Wo1_h, H_ * H_);
  cvt_f32_f16_kernel<<<blocks(H_ * H_), TB, 0, stream>>>(W_out2, Wo2_h, H_ * H_);

  // --- clinical MLPs: hidden0 and cln_msg (M=16, tiny -> fp32 VALU) ---
  linear_f32_kernel<<<blocks(B_ * H_), TB, 0, stream>>>(clinical, Wc1, bc1, t1, B_, H_, C_, 1);
  linear_f32_kernel<<<blocks(B_ * 2 * H_), TB, 0, stream>>>(t1, Wc2, bc2, t2, B_, 2 * H_, H_, 1);
  linear_f32_kernel<<<blocks(B_ * A_ * H_), TB, 0, stream>>>(t2, Wc3, bc3, hid, B_, A_ * H_, 2 * H_, 0);
  linear_f32_kernel<<<blocks(B_ * H_), TB, 0, stream>>>(clinical, Wm1, bm1, t1, B_, H_, C_, 1);
  linear_f32_kernel<<<blocks(B_ * 2 * H_), TB, 0, stream>>>(t1, Wm2, bm2, t2, B_, 2 * H_, H_, 1);
  linear_f32_kernel<<<blocks(B_ * A_ * H_), TB, 0, stream>>>(t2, Wm3, bm3, cln, B_, A_ * H_, 2 * H_, 0);
  cvt_f32_f16_kernel<<<blocks(MN * H_), TB, 0, stream>>>(hid, hid_h, MN * H_);

  // --- x0 and the t=0 output slice ---
  init_x_kernel<<<blocks(MN * F_), TB, 0, stream>>>(data, xcur, out);

  const dim3 gridEdge(ME / 64, H_ / 32, K_);   // GEMM1: z over K edge types
  const dim3 gridMsg2(ME / 64, H_ / 32, 1);
  const dim3 gridNode(MN / 64, H_ / 32, 1);

  for (int t = 1; t < T_; ++t) {
    // gather recv||send features (one-hot matmuls done analytically)
    premsg_kernel<<<blocks(ME * 2 * H_), TB, 0, stream>>>(hid_h, pmsg);

    // h1[k] = relu(pre_msg @ W_msg1[k]^T + b_msg1[k])   [WMMA, KD=512]
    wmma_gemm_kernel<2 * H_><<<gridEdge, TB, 0, stream>>>(
        pmsg, (size_t)0, W1h, (size_t)H_ * 2 * H_, b_msg1, (size_t)H_,
        h1, nullptr, (size_t)ME * H_, H_, 1);

    // all_msgs = sum_k rel_type[:,k] * relu(h1[k] @ W_msg2[k]^T + b_msg2[k])
    edge_msg2_kernel<<<gridMsg2, TB, 0, stream>>>(h1, W2h, b_msg2, rel_type, msgs);

    // agg = scatter-add over receivers (contiguous) + cln_msg -> f16
    aggregate_kernel<<<blocks(MN * H_), TB, 0, stream>>>(msgs, cln, aggh);

    // hr_lin / hn_lin = agg @ W_{hr,hn}^T + b   [WMMA, KD=256]
    wmma_gemm_kernel<H_><<<gridNode, TB, 0, stream>>>(
        aggh, (size_t)0, Whr_h, (size_t)0, b_hr, (size_t)0,
        nullptr, hrl, (size_t)0, H_, 0);
    wmma_gemm_kernel<H_><<<gridNode, TB, 0, stream>>>(
        aggh, (size_t)0, Whn_h, (size_t)0, b_hn, (size_t)0,
        nullptr, hnl, (size_t)0, H_, 0);

    // fused GRU cell: updates hid (f32 carry) and hid_h (f16 for GEMMs)
    gru_kernel<<<blocks(MN * H_), TB, 0, stream>>>(
        xcur, hrl, hnl, W_in_r, b_in_r, W_in_z, b_in_z, W_in_n, b_in_n, hid, hid_h);

    // output MLP: o1 = relu(hid @ W_out1^T + b), o2 = relu(o1 @ W_out2^T + b)
    wmma_gemm_kernel<H_><<<gridNode, TB, 0, stream>>>(
        hid_h, (size_t)0, Wo1_h, (size_t)0, b_out1, (size_t)0,
        o1, nullptr, (size_t)0, H_, 1);
    wmma_gemm_kernel<H_><<<gridNode, TB, 0, stream>>>(
        o1, (size_t)0, Wo2_h, (size_t)0, b_out2, (size_t)0,
        o2, nullptr, (size_t)0, H_, 1);

    // pred = x + o2 @ W_out3^T + b_out3 ; write out[:,:,t,:], advance x
    pred_kernel<<<blocks(MN * F_), TB, 0, stream>>>(o2, W_out3, b_out3, xcur, out, t);
  }
}
